// GCN_74741020885173
// MI455X (gfx1250) — matrix operand order repaired and checked by reading, not compiled, not run on hardware
//
#include <hip/hip_runtime.h>
#include <hip/hip_bf16.h>

typedef __attribute__((ext_vector_type(16))) __bf16 v16bf;
typedef __attribute__((ext_vector_type(8)))  float  v8f;

#define F 128   // feature width (F_IN == H == 128)
#define C_OUT 16
#define NUM_GRAPHS 512

// ---------------- utility kernels ----------------

__global__ void fill_f32(float* __restrict__ p, float v, int n) {
    int i = blockIdx.x * blockDim.x + threadIdx.x;
    if (i < n) p[i] = v;
}

__global__ void deg_scatter(const int* __restrict__ col, float* __restrict__ deg, int E) {
    int e = blockIdx.x * blockDim.x + threadIdx.x;
    if (e < E) atomicAdd(&deg[col[e]], 1.0f);
}

__global__ void rsqrt_inplace(float* __restrict__ d, int n) {
    int i = blockIdx.x * blockDim.x + threadIdx.x;
    if (i < n) {
        float v = d[i];
        d[i] = (v > 0.0f) ? rsqrtf(v) : 0.0f;
    }
}

__global__ void edge_norm(const int* __restrict__ row, const int* __restrict__ col,
                          const float* __restrict__ dinv, float* __restrict__ norm, int E) {
    int e = blockIdx.x * blockDim.x + threadIdx.x;
    if (e < E) norm[e] = dinv[row[e]] * dinv[col[e]];
}

// ---------------- weight pre-pack: f32 [128][128] -> bf16 WMMA B-fragments ----
// Fragment layout: P[(kc*8 + nt)*32 + lane] is the v16bf this lane feeds to
// v_wmma for K-chunk kc, N-tile nt.
//   n  = nt*16 + (lane & 15)
//   k  = kc*32 + (lane>>4)*16 + j        (j = 0..15)
__global__ void pack_w(const float* __restrict__ W, v16bf* __restrict__ P) {
    int t = blockIdx.x * blockDim.x + threadIdx.x;    // 1024 threads total
    if (t >= 1024) return;
    int lane = t & 31;
    int f    = t >> 5;            // kc*8 + nt
    int kc   = f >> 3;
    int nt   = f & 7;
    int n    = nt * 16 + (lane & 15);
    int kb   = kc * 32 + (lane >> 4) * 16;
    v16bf b;
#pragma unroll
    for (int j = 0; j < 16; ++j) b[j] = (__bf16)W[(size_t)(kb + j) * F + n];
    P[t] = b;
}

// ---------------- dense transform: H = [relu](X) @ W  (WMMA bf16 -> f32) -----
// One wave computes a 16x128 tile. Block = 4 waves = 64 rows.
// OOB rows are handled by clamping the A row index (duplicate data only feeds
// D rows that are never stored), so all A loads are unconditional b128s.
// RELU is a compile-time flag so no per-element select chains are emitted.

template <int RELU>
__global__ __launch_bounds__(128)
void gemm_wmma_bf16(const float* __restrict__ X, const v16bf* __restrict__ Wp,
                    float* __restrict__ H, int N) {
    const int lane    = threadIdx.x & 31;
    const int wave    = threadIdx.x >> 5;
    const int rowBase = blockIdx.x * 64 + wave * 16;
    const int m       = lane & 15;
    const int half    = lane >> 4;

    v8f acc[8];
#pragma unroll
    for (int t = 0; t < 8; ++t) {
        v8f z = {0.f, 0.f, 0.f, 0.f, 0.f, 0.f, 0.f, 0.f};
        acc[t] = z;
    }

    int r = rowBase + m;
    if (r >= N) r = N - 1;                       // clamp: safe duplicate row
    const float* xrow = X + (size_t)r * F;

#pragma unroll
    for (int kc = 0; kc < 4; ++kc) {
        // --- A fragment: 16x32 slice of X, bf16; optional fused relu ---
        const int kbA = kc * 32 + half * 16;
        const float* xs = xrow + kbA;
        float4 p0 = *(const float4*)(xs);        // k = kbA + 0..3
        float4 p1 = *(const float4*)(xs + 4);    // k = kbA + 4..7
        float4 p2 = *(const float4*)(xs + 8);    // k = kbA + 8..11
        float4 p3 = *(const float4*)(xs + 12);   // k = kbA + 12..15
        float va[16] = { p0.x, p0.y, p0.z, p0.w, p1.x, p1.y, p1.z, p1.w,
                         p2.x, p2.y, p2.z, p2.w, p3.x, p3.y, p3.z, p3.w };
        v16bf a;
#pragma unroll
        for (int j = 0; j < 16; ++j) {
            float v = va[j];
            if (RELU) v = fmaxf(v, 0.0f);
            a[j] = (__bf16)v;
        }

        // --- sweep the 8 column tiles; B fragments are pre-packed ---
#pragma unroll
        for (int nt = 0; nt < 8; ++nt) {
            v16bf b = Wp[(kc * 8 + nt) * 32 + lane];
            acc[nt] = __builtin_amdgcn_wmma_f32_16x16x32_bf16(
                false, a, false, b, (short)0, acc[nt], false, false);
        }
    }

    // --- store D: VGPR v -> row rowBase + v + 8*half, col nt*16 + m ---
    const int mOff = half * 8;
    if (rowBase + 16 <= N) {
        // fast path: full tile, unconditional coalesced stores
#pragma unroll
        for (int v = 0; v < 8; ++v) {
            float* dst = H + (size_t)(rowBase + mOff + v) * F + m;
#pragma unroll
            for (int nt = 0; nt < 8; ++nt) dst[nt * 16] = acc[nt][v];
        }
    } else {
        // tail tile: per-row guard
#pragma unroll
        for (int v = 0; v < 8; ++v) {
            const int rr = rowBase + mOff + v;
            if (rr < N) {
                float* dst = H + (size_t)rr * F + m;
#pragma unroll
                for (int nt = 0; nt < 8; ++nt) dst[nt * 16] = acc[nt][v];
            }
        }
    }
}

// ---------------- aggregation ----------------
// out[i,:] = h[i,:]*dinv[i]^2 + b   (self-loop term + bias; full overwrite)
__global__ void agg_init(const float* __restrict__ h, const float* __restrict__ dinv,
                         const float* __restrict__ bias, float* __restrict__ out, int N) {
    int idx  = blockIdx.x * blockDim.x + threadIdx.x;
    int i    = idx >> 5;
    int lane = idx & 31;
    if (i >= N) return;
    float sn = dinv[i]; sn *= sn;
    float4 hv = *((const float4*)(h + (size_t)i * F) + lane);
    float4 bv = *((const float4*)bias + lane);
    float4 o  = { hv.x * sn + bv.x, hv.y * sn + bv.y,
                  hv.z * sn + bv.z, hv.w * sn + bv.w };
    *((float4*)(out + (size_t)i * F) + lane) = o;
}

// out[col[e],:] += h[row[e],:] * norm[e]  — one wave per edge, float4 per lane
__global__ void edge_scatter(const float* __restrict__ h, const int* __restrict__ row,
                             const int* __restrict__ col, const float* __restrict__ norm,
                             float* __restrict__ out, int E) {
    long long idx = (long long)blockIdx.x * blockDim.x + threadIdx.x;
    int e    = (int)(idx >> 5);
    int lane = (int)(idx & 31);
    if (e >= E) return;
    int   rs = row[e];
    int   cs = col[e];
    float nv = norm[e];
    float4 v = *((const float4*)(h + (size_t)rs * F) + lane);
    float* op = out + (size_t)cs * F + lane * 4;
    atomicAdd(op + 0, v.x * nv);
    atomicAdd(op + 1, v.y * nv);
    atomicAdd(op + 2, v.z * nv);
    atomicAdd(op + 3, v.w * nv);
}

// ---------------- pooling + classifier ----------------

__global__ void pool_scatter(const float* __restrict__ h, const int* __restrict__ batch,
                             float* __restrict__ pooled, float* __restrict__ cnt, int N) {
    int idx  = blockIdx.x * blockDim.x + threadIdx.x;
    int i    = idx >> 5;
    int lane = idx & 31;
    if (i >= N) return;
    int g = batch[i];
    float4 v = *((const float4*)(h + (size_t)i * F) + lane);
    float* pp = pooled + (size_t)g * F + lane * 4;
    atomicAdd(pp + 0, v.x);
    atomicAdd(pp + 1, v.y);
    atomicAdd(pp + 2, v.z);
    atomicAdd(pp + 3, v.w);
    if (lane == 0) atomicAdd(&cnt[g], 1.0f);
}

__global__ void final_linear(const float* __restrict__ pooled, const float* __restrict__ cnt,
                             const float* __restrict__ W, const float* __restrict__ b,
                             float* __restrict__ out) {
    int t = blockIdx.x * blockDim.x + threadIdx.x;
    if (t >= NUM_GRAPHS * C_OUT) return;
    int g = t >> 4;
    int c = t & 15;
    float inv = 1.0f / fmaxf(cnt[g], 1.0f);
    const float* pg = pooled + (size_t)g * F;
    float s = 0.0f;
#pragma unroll 8
    for (int k = 0; k < F; ++k) s += pg[k] * W[k * C_OUT + c];
    out[t] = s * inv + b[c];
}

// ---------------- host launcher ----------------

extern "C" void kernel_launch(void* const* d_in, const int* in_sizes, int n_in,
                              void* d_out, int out_size, void* d_ws, size_t ws_size,
                              hipStream_t stream) {
    const float* x     = (const float*)d_in[0];
    const int*   edge  = (const int*)d_in[1];
    const int*   batch = (const int*)d_in[2];
    const float* W1    = (const float*)d_in[3];
    const float* b1    = (const float*)d_in[4];
    const float* W2    = (const float*)d_in[5];
    const float* b2    = (const float*)d_in[6];
    const float* W3    = (const float*)d_in[7];
    const float* b3    = (const float*)d_in[8];
    const float* linW  = (const float*)d_in[9];
    const float* linb  = (const float*)d_in[10];

    const int N = in_sizes[0] / F;
    const int E = in_sizes[1] / 2;
    const int* row = edge;
    const int* col = edge + E;

    // workspace carve-out
    unsigned char* ws = (unsigned char*)d_ws;
    size_t off = 0;
    auto carve = [&](size_t bytes) -> void* {
        void* p = (void*)(ws + off);
        off = (off + bytes + 255) & ~(size_t)255;
        return p;
    };
    float* dinv   = (float*)carve((size_t)N * 4);
    float* norm   = (float*)carve((size_t)E * 4);
    float* bufA   = (float*)carve((size_t)N * F * 4);   // h = in @ W
    float* bufB   = (float*)carve((size_t)N * F * 4);   // aggregated layer output
    float* pooled = (float*)carve((size_t)NUM_GRAPHS * F * 4);
    float* cnt    = (float*)carve((size_t)NUM_GRAPHS * 4);
    v16bf* W1p    = (v16bf*)carve(1024 * 32);           // packed bf16 fragments
    v16bf* W2p    = (v16bf*)carve(1024 * 32);
    v16bf* W3p    = (v16bf*)carve(1024 * 32);
    (void)ws_size; (void)n_in;

    const int TB = 256;
    const int gN    = (N + TB - 1) / TB;
    const int gE    = (E + TB - 1) / TB;
    const int gN32  = (int)(((long long)N * 32 + TB - 1) / TB);
    const int gE32  = (int)(((long long)E * 32 + TB - 1) / TB);
    const int gGemm = (N + 63) / 64;

    // pack weights into WMMA fragment layout (tiny, L2-resident afterwards)
    pack_w<<<4, 256, 0, stream>>>(W1, W1p);
    pack_w<<<4, 256, 0, stream>>>(W2, W2p);
    pack_w<<<4, 256, 0, stream>>>(W3, W3p);

    // degree normalization: deg starts at 1 (self loop), + in-degree over col
    fill_f32<<<gN, TB, 0, stream>>>(dinv, 1.0f, N);
    deg_scatter<<<gE, TB, 0, stream>>>(col, dinv, E);
    rsqrt_inplace<<<gN, TB, 0, stream>>>(dinv, N);
    edge_norm<<<gE, TB, 0, stream>>>(row, col, dinv, norm, E);

    // layer 1
    gemm_wmma_bf16<0><<<gGemm, 128, 0, stream>>>(x, W1p, bufA, N);
    agg_init<<<gN32, TB, 0, stream>>>(bufA, dinv, b1, bufB, N);
    edge_scatter<<<gE32, TB, 0, stream>>>(bufA, row, col, norm, bufB, E);

    // layer 2 (relu of layer-1 output fused into GEMM A-load)
    gemm_wmma_bf16<1><<<gGemm, 128, 0, stream>>>(bufB, W2p, bufA, N);
    agg_init<<<gN32, TB, 0, stream>>>(bufA, dinv, b2, bufB, N);
    edge_scatter<<<gE32, TB, 0, stream>>>(bufA, row, col, norm, bufB, E);

    // layer 3 (relu of layer-2 output fused; no relu after)
    gemm_wmma_bf16<1><<<gGemm, 128, 0, stream>>>(bufB, W3p, bufA, N);
    agg_init<<<gN32, TB, 0, stream>>>(bufA, dinv, b3, bufB, N);
    edge_scatter<<<gE32, TB, 0, stream>>>(bufA, row, col, norm, bufB, E);

    // global mean pool + classifier
    fill_f32<<<(NUM_GRAPHS * F + TB - 1) / TB, TB, 0, stream>>>(pooled, 0.0f, NUM_GRAPHS * F);
    fill_f32<<<(NUM_GRAPHS + TB - 1) / TB, TB, 0, stream>>>(cnt, 0.0f, NUM_GRAPHS);
    pool_scatter<<<gN32, TB, 0, stream>>>(bufB, batch, pooled, cnt, N);
    final_linear<<<(NUM_GRAPHS * C_OUT + TB - 1) / TB, TB, 0, stream>>>(
        pooled, cnt, linW, linb, (float*)d_out);
    (void)out_size;
}
